// MarketLSTM_64407329571077
// MI455X (gfx1250) — compile-verified, matrix operand order checked by reading, not run on hardware
//
#include <hip/hip_runtime.h>

typedef __attribute__((ext_vector_type(16))) __bf16 v16bf;
typedef __attribute__((ext_vector_type(8)))  __bf16 v8bf;
typedef __attribute__((ext_vector_type(8)))  float  v8f;

#define TT 512
#define DD 5
#define HH 50

__device__ __forceinline__ float tanh_fast(float x) {
#if __has_builtin(__builtin_amdgcn_tanhf)
  return __builtin_amdgcn_tanhf(x);     // gfx1250 v_tanh_f32
#else
  return __builtin_tanhf(x);
#endif
}
__device__ __forceinline__ float sigm_fast(float x) {
  return __builtin_fmaf(0.5f, tanh_fast(0.5f * x), 0.5f);
}

// K position inside one 32-wide K chunk for half-slot h of `lane`
// (16-bit WMMA A/B operand layout, wave32)
__device__ __forceinline__ int kslot(int lane, int h) {
  return (h & 7) + ((h >> 3) << 4) + ((lane >> 4) << 3);
}

// layer0 combined weight W[n][k], n = 64*gate + j, K = [x(5) | h0(50) | pad]
__device__ __forceinline__ float w0_fetch(const float* __restrict__ Wih,
                                          const float* __restrict__ Whh,
                                          int gate, int j, int k) {
  if (j >= HH) return 0.0f;
  int row = gate * HH + j;
  if (k < DD)      return Wih[row * DD + k];
  if (k < DD + HH) return Whh[row * HH + (k - DD)];
  return 0.0f;
}
// layer1 combined weight, K = [h0(50) | h1(50) | pad]
__device__ __forceinline__ float w1_fetch(const float* __restrict__ Wih,
                                          const float* __restrict__ Whh,
                                          int gate, int j, int k) {
  if (j >= HH) return 0.0f;
  int row = gate * HH + j;
  if (k < HH)     return Wih[row * HH + k];
  if (k < 2 * HH) return Whh[row * HH + (k - HH)];
  return 0.0f;
}

// Load one v16bf A-operand chunk: halves 0..7 at p[0..7], 8..15 at p[16..23]
__device__ __forceinline__ v16bf load_a16(const __bf16* p) {
  v8bf lo = *(const v8bf*)(p);
  v8bf hi = *(const v8bf*)(p + 16);
  return __builtin_shufflevector(lo, hi, 0, 1, 2, 3, 4, 5, 6, 7,
                                 8, 9, 10, 11, 12, 13, 14, 15);
}

__global__ __launch_bounds__(128) void lstm_fused(
    const float* __restrict__ x,
    const float* __restrict__ Wih0, const float* __restrict__ Whh0,
    const float* __restrict__ bih0, const float* __restrict__ bhh0,
    const float* __restrict__ Wih1, const float* __restrict__ Whh1,
    const float* __restrict__ bih1, const float* __restrict__ bhh1,
    const float* __restrict__ Wlin, const float* __restrict__ blin,
    float* __restrict__ out)
{
  __shared__ __align__(16) __bf16 sA0[16][64];    // [x_t | h0 | pad]  bf16
  __shared__ __align__(16) __bf16 sA1[16][128];   // [h0 | h1 | pad]   bf16
  __shared__ float sP[4][16];

  const int tid  = threadIdx.x;
  const int lane = tid & 31;
  const int wv   = tid >> 5;            // wave 0..3 -> hidden cols 16*wv..
  const int col  = lane & 15;
  const int hsel = (lane >> 4) << 3;    // 0 or 8: row offset (D) / K-half (A)
  const int j    = wv * 16 + col;       // hidden unit this lane handles
  const int brow = blockIdx.x * 16;     // batch tile base

  // ---- one-time: gather weights into registers in WMMA B layout (bf16) ----
  v16bf B0t[4][2];                      // [gate][kchunk], layer 0
  v16bf B1t[4][4];                      // [gate][kchunk], layer 1
#pragma unroll
  for (int a = 0; a < 4; ++a) {
#pragma unroll
    for (int c = 0; c < 2; ++c) {
      v16bf v;
#pragma unroll
      for (int h = 0; h < 16; ++h)
        v[h] = (__bf16)w0_fetch(Wih0, Whh0, a, j, c * 32 + kslot(lane, h));
      B0t[a][c] = v;
    }
#pragma unroll
    for (int c = 0; c < 4; ++c) {
      v16bf v;
#pragma unroll
      for (int h = 0; h < 16; ++h)
        v[h] = (__bf16)w1_fetch(Wih1, Whh1, a, j, c * 32 + kslot(lane, h));
      B1t[a][c] = v;
    }
  }

  float bias0v[4], bias1v[4];
#pragma unroll
  for (int a = 0; a < 4; ++a) {
    if (j < HH) {
      bias0v[a] = bih0[a * HH + j] + bhh0[a * HH + j];
      bias1v[a] = bih1[a * HH + j] + bhh1[a * HH + j];
    } else {
      bias0v[a] = 0.0f;
      bias1v[a] = 0.0f;
    }
  }

  // zero staging (padded cols stay zero forever)
  for (int i = tid; i < 16 * 64; i += 128)  (&sA0[0][0])[i] = (__bf16)0.0f;
  for (int i = tid; i < 16 * 128; i += 128) (&sA1[0][0])[i] = (__bf16)0.0f;

  float c0v[8], c1v[8], h1v[8];
#pragma unroll
  for (int r = 0; r < 8; ++r) { c0v[r] = 0.0f; c1v[r] = 0.0f; h1v[r] = 0.0f; }

  for (int t = 0; t < TT; ++t) {
    __syncthreads();                          // prev-step sA1 reads complete

    if (wv == 0 && lane < 16) {               // stage x_t
      const float* xp = x + ((size_t)(brow + lane) * TT + t) * DD;
#pragma unroll
      for (int d = 0; d < DD; ++d) sA0[lane][d] = (__bf16)xp[d];
    }
    if (j < HH) {                             // stage h1_{t-1}
#pragma unroll
      for (int r = 0; r < 8; ++r) sA1[hsel + r][HH + j] = (__bf16)h1v[r];
    }
    __syncthreads();                          // x_t, h1_{t-1} visible

    // -------- layer 0: gates = [x_t | h0] @ W0 + b0 --------
    v16bf a0c0 = load_a16(&sA0[col][0]  + hsel);
    v16bf a0c1 = load_a16(&sA0[col][32] + hsel);
    v8f acc[4];
#pragma unroll
    for (int a = 0; a < 4; ++a) {
      v8f d;
#pragma unroll
      for (int r = 0; r < 8; ++r) d[r] = bias0v[a];
      d = __builtin_amdgcn_wmma_f32_16x16x32_bf16(false, a0c0, false, B0t[a][0],
                                                  (short)0, d, false, false);
      d = __builtin_amdgcn_wmma_f32_16x16x32_bf16(false, a0c1, false, B0t[a][1],
                                                  (short)0, d, false, false);
      acc[a] = d;
    }
    float h0v[8];
#pragma unroll
    for (int r = 0; r < 8; ++r) {
      float ig = sigm_fast(acc[0][r]);
      float fg = sigm_fast(acc[1][r]);
      float gg = tanh_fast(acc[2][r]);
      float og = sigm_fast(acc[3][r]);
      float c  = __builtin_fmaf(fg, c0v[r], ig * gg);
      c0v[r]   = c;
      h0v[r]   = og * tanh_fast(c);
    }
    __syncthreads();                          // all waves done reading sA0/sA1 h0 region

    if (j < HH) {                             // stage h0_t for both consumers
#pragma unroll
      for (int r = 0; r < 8; ++r) {
        __bf16 hb = (__bf16)h0v[r];
        sA0[hsel + r][DD + j] = hb;
        sA1[hsel + r][j]      = hb;
      }
    }
    __syncthreads();                          // h0_t visible

    // -------- layer 1: gates = [h0_t | h1_{t-1}] @ W1 + b1 --------
    v16bf a1c0 = load_a16(&sA1[col][0]  + hsel);
    v16bf a1c1 = load_a16(&sA1[col][32] + hsel);
    v16bf a1c2 = load_a16(&sA1[col][64] + hsel);
    v16bf a1c3 = load_a16(&sA1[col][96] + hsel);
#pragma unroll
    for (int a = 0; a < 4; ++a) {
      v8f d;
#pragma unroll
      for (int r = 0; r < 8; ++r) d[r] = bias1v[a];
      d = __builtin_amdgcn_wmma_f32_16x16x32_bf16(false, a1c0, false, B1t[a][0],
                                                  (short)0, d, false, false);
      d = __builtin_amdgcn_wmma_f32_16x16x32_bf16(false, a1c1, false, B1t[a][1],
                                                  (short)0, d, false, false);
      d = __builtin_amdgcn_wmma_f32_16x16x32_bf16(false, a1c2, false, B1t[a][2],
                                                  (short)0, d, false, false);
      d = __builtin_amdgcn_wmma_f32_16x16x32_bf16(false, a1c3, false, B1t[a][3],
                                                  (short)0, d, false, false);
      acc[a] = d;
    }
#pragma unroll
    for (int r = 0; r < 8; ++r) {
      float ig = sigm_fast(acc[0][r]);
      float fg = sigm_fast(acc[1][r]);
      float gg = tanh_fast(acc[2][r]);
      float og = sigm_fast(acc[3][r]);
      float c  = __builtin_fmaf(fg, c1v[r], ig * gg);
      c1v[r]   = c;
      h1v[r]   = og * tanh_fast(c);
    }
  }

  // -------- deterministic epilogue: out = h1_T @ Wlin^T + blin --------
  float wl = (j < HH) ? Wlin[j] : 0.0f;
  float part[8];
#pragma unroll
  for (int r = 0; r < 8; ++r) part[r] = h1v[r] * wl;
#pragma unroll
  for (int m = 1; m <= 8; m <<= 1) {
#pragma unroll
    for (int r = 0; r < 8; ++r) part[r] += __shfl_xor(part[r], m, 32);
  }
  __syncthreads();
  if (col == 0) {
#pragma unroll
    for (int r = 0; r < 8; ++r) sP[wv][hsel + r] = part[r];
  }
  __syncthreads();
  if (tid < 16) {
    out[brow + tid] = sP[0][tid] + sP[1][tid] + sP[2][tid] + sP[3][tid] + blin[0];
  }
}

extern "C" void kernel_launch(void* const* d_in, const int* in_sizes, int n_in,
                              void* d_out, int out_size, void* d_ws, size_t ws_size,
                              hipStream_t stream) {
  const float* x    = (const float*)d_in[0];
  const float* Wih0 = (const float*)d_in[1];
  const float* Whh0 = (const float*)d_in[2];
  const float* bih0 = (const float*)d_in[3];
  const float* bhh0 = (const float*)d_in[4];
  const float* Wih1 = (const float*)d_in[5];
  const float* Whh1 = (const float*)d_in[6];
  const float* bih1 = (const float*)d_in[7];
  const float* bhh1 = (const float*)d_in[8];
  const float* Wlin = (const float*)d_in[9];
  const float* blin = (const float*)d_in[10];
  float* out = (float*)d_out;

  int B = in_sizes[0] / (TT * DD);          // 4096
  int blocks = (B + 15) / 16;               // 256 blocks x 4 waves
  lstm_fused<<<blocks, 128, 0, stream>>>(x, Wih0, Whh0, bih0, bhh0,
                                         Wih1, Whh1, bih1, bhh1,
                                         Wlin, blin, out);
}
